// Attention_50242527428847
// MI455X (gfx1250) — matrix operand order repaired and checked by reading, not compiled, not run on hardware
//
#include <hip/hip_runtime.h>

// Problem constants (match reference)
#define TD    128
#define TE    128
#define BB    8
#define ENC_H 512
#define DEC_H 1024
#define D_IN  256

typedef __attribute__((ext_vector_type(16))) __bf16 v16bf;
typedef __attribute__((ext_vector_type(8)))  __bf16 v8bf;
typedef __attribute__((ext_vector_type(8)))  float  v8f;
typedef unsigned short us16;

// round-to-nearest-even f32 -> bf16 (bit trick; header-independent)
static __device__ __forceinline__ us16 f2bf(float f) {
  unsigned int u = __float_as_uint(f);
  u += 0x7FFFu + ((u >> 16) & 1u);
  return (us16)(u >> 16);
}

// ---------------------------------------------------------------------------
// Fused pack kernel: one launch converts all three operands to bf16.
//   range 0: decoder_states (TD*B, DEC_H) -> Abf       [1,048,576 elems]
//   range 1: W (ENC_H, DEC_H)             -> Wbf       [  524,288 elems]
//   range 2: enc_in (TE,B,D_IN) -> EinT [b][d][s]      [  262,144 elems]
// ---------------------------------------------------------------------------
#define N_A   (TD * BB * DEC_H)
#define N_W   (ENC_H * DEC_H)
#define N_E   (BB * D_IN * TE)

__global__ void __launch_bounds__(256)
pack_all(const float* __restrict__ dec_st, const float* __restrict__ W,
         const float* __restrict__ enc_in, us16* __restrict__ Abf,
         us16* __restrict__ Wbf, us16* __restrict__ EinT) {
  int i = blockIdx.x * blockDim.x + threadIdx.x;
  if (i < N_A) {
    Abf[i] = f2bf(dec_st[i]);
  } else if (i < N_A + N_W) {
    int j = i - N_A;
    Wbf[j] = f2bf(W[j]);
  } else if (i < N_A + N_W + N_E) {
    int j = i - (N_A + N_W);          // j = (b*D_IN + d)*TE + s
    int s  = j & (TE - 1);
    int bd = j >> 7;
    int d  = bd & (D_IN - 1);
    int b  = bd >> 8;
    EinT[j] = f2bf(enc_in[(s * BB + b) * D_IN + d]);
  }
}

// ---------------------------------------------------------------------------
// Register-blocked WMMA micro-kernel: one wave computes a 16x64 output strip
// (4 accumulators share each A operand). All 4 B tiles are loaded into
// DISTINCT registers before any WMMA consumes them, so loads clause together
// and WMMAs retire against partial loadcnt instead of draining the pipe
// (avoids the round-2 `s_wait_loadcnt 0x0` WAR serialization).
// A per-lane layout: lane = m, 16 contiguous K from half*16.
// B: lane = n, 16 contiguous K from half*16.
// ---------------------------------------------------------------------------
__global__ void __launch_bounds__(32)
gemm_dec(const us16* __restrict__ Abf, const us16* __restrict__ Wbf,
         float* __restrict__ dec) {
  int lane = threadIdx.x;
  int half = lane >> 4, l16 = lane & 15;
  int mT = blockIdx.x;        // 0..63  (16-row strip of M=1024)
  int nG = blockIdx.y;        // 0..7   (64-col group of N=512)
  const us16* arow = Abf + (mT * 16 + l16) * DEC_H + half * 16;
  const us16* brow = Wbf + (nG * 64 + l16) * DEC_H + half * 16;
  v8f c[4] = {{}, {}, {}, {}};
  for (int kk = 0; kk < DEC_H; kk += 32) {
    union { v16bf v; v8bf h[2]; } a;
    a.h[0] = *(const v8bf*)(arow + kk);
    a.h[1] = *(const v8bf*)(arow + kk + 8);
    v16bf bm[4];
#pragma unroll
    for (int j = 0; j < 4; ++j)
      bm[j] = *(const v16bf*)(brow + j * 16 * DEC_H + kk);
#pragma unroll
    for (int j = 0; j < 4; ++j)
      c[j] = __builtin_amdgcn_wmma_f32_16x16x32_bf16(
          false, a.v, false, bm[j], (short)0, c[j], false, false);
  }
  // C/D layout: lanes0-15: VGPR r -> M=r, N=lane; lanes16-31: M=r+8, N=lane-16
#pragma unroll
  for (int j = 0; j < 4; ++j)
#pragma unroll
    for (int r = 0; r < 8; ++r)
      dec[(mT * 16 + half * 8 + r) * ENC_H + nG * 64 + j * 16 + l16] = c[j][r];
}

// ---------------------------------------------------------------------------
// Fused column-softmax over s (axis=1) + wsum[t,s,b] = sum_e weights
//   scores[s,e] = enc_states[s,b,e] * dec[t,b,e]
// grid(TD, B), 256 threads. Pass1/2: two columns per thread (coalesced over e).
// Pass3: one wave per row s, lanes stride e, wave32 shuffle reduction.
// ---------------------------------------------------------------------------
__global__ void __launch_bounds__(256)
softmax_wsum(const float* __restrict__ enc_st, const float* __restrict__ dec,
             us16* __restrict__ wsum) {
  __shared__ float s_dec[ENC_H];
  __shared__ float s_m[ENC_H];
  __shared__ float s_rZ[ENC_H];
  __shared__ float s_w[TE];
  int t = blockIdx.x, b = blockIdx.y, tid = threadIdx.x;
  const float* drow = dec + (t * BB + b) * ENC_H;
  s_dec[tid]       = drow[tid];
  s_dec[tid + 256] = drow[tid + 256];
  __syncthreads();
#pragma unroll
  for (int cc = 0; cc < 2; ++cc) {
    int e = tid + cc * 256;
    float dv = s_dec[e];
    const float* col = enc_st + b * ENC_H + e;   // stride B*ENC_H over s
    float m = -3.4e38f;
    for (int s = 0; s < TE; ++s)
      m = fmaxf(m, col[s * BB * ENC_H] * dv);
    float Z = 0.f;
    for (int s = 0; s < TE; ++s)
      Z += __expf(col[s * BB * ENC_H] * dv - m);
    s_m[e]  = m;
    s_rZ[e] = 1.0f / Z;
  }
  __syncthreads();
  int wave = tid >> 5, lane = tid & 31;
  for (int s = wave; s < TE; s += 8) {
    const float* row = enc_st + (s * BB + b) * ENC_H;
    float acc = 0.f;
    for (int e = lane; e < ENC_H; e += 32)
      acc += __expf(row[e] * s_dec[e] - s_m[e]) * s_rZ[e];
#pragma unroll
    for (int off = 16; off; off >>= 1)
      acc += __shfl_xor(acc, off, 32);
    if (lane == 0) s_w[s] = acc;
  }
  __syncthreads();
  if (tid < TE) wsum[(b * TD + t) * TE + tid] = f2bf(s_w[tid]);
}

// ---------------------------------------------------------------------------
// GEMM2 (per b): out[t,b,d] = sum_s wsum_bf16[b][t][s] * EinT[b][d][s]
// grid(TD/16, D_IN/64, B): one wave computes 16x64, K=TE=128 -> 16 WMMAs
// ---------------------------------------------------------------------------
__global__ void __launch_bounds__(32)
gemm_out(const us16* __restrict__ Ws, const us16* __restrict__ EinT,
         float* __restrict__ out) {
  int lane = threadIdx.x;
  int half = lane >> 4, l16 = lane & 15;
  int mT = blockIdx.x, nG = blockIdx.y, b = blockIdx.z;
  const us16* arow = Ws   + (b * TD   + mT * 16 + l16) * TE + half * 16;
  const us16* brow = EinT + (b * D_IN + nG * 64 + l16) * TE + half * 16;
  v8f c[4] = {{}, {}, {}, {}};
#pragma unroll
  for (int kk = 0; kk < TE; kk += 32) {
    union { v16bf v; v8bf h[2]; } a;
    a.h[0] = *(const v8bf*)(arow + kk);
    a.h[1] = *(const v8bf*)(arow + kk + 8);
    v16bf bm[4];
#pragma unroll
    for (int j = 0; j < 4; ++j)
      bm[j] = *(const v16bf*)(brow + j * 16 * TE + kk);
#pragma unroll
    for (int j = 0; j < 4; ++j)
      c[j] = __builtin_amdgcn_wmma_f32_16x16x32_bf16(
          false, a.v, false, bm[j], (short)0, c[j], false, false);
  }
#pragma unroll
  for (int j = 0; j < 4; ++j)
#pragma unroll
    for (int r = 0; r < 8; ++r) {
      int t = mT * 16 + half * 8 + r;
      int d = nG * 64 + j * 16 + l16;
      out[(t * BB + b) * D_IN + d] = c[j][r];
    }
}

// ---------------------------------------------------------------------------
extern "C" void kernel_launch(void* const* d_in, const int* in_sizes, int n_in,
                              void* d_out, int out_size, void* d_ws, size_t ws_size,
                              hipStream_t stream) {
  (void)in_sizes; (void)n_in; (void)out_size; (void)ws_size;
  const float* enc_in = (const float*)d_in[0];   // (TE,B,D_IN)
  const float* enc_st = (const float*)d_in[1];   // (TE,B,ENC_H)
  const float* dec_st = (const float*)d_in[2];   // (TD,B,DEC_H)
  const float* W      = (const float*)d_in[3];   // (ENC_H,DEC_H)
  float* out = (float*)d_out;                    // (TD,B,D_IN)

  char* ws = (char*)d_ws;
  us16*  Abf  = (us16*)(ws + 0);                              // 2.00 MB
  us16*  Wbf  = (us16*)(ws + (2u << 20));                     // 1.00 MB
  us16*  EinT = (us16*)(ws + (3u << 20));                     // 0.50 MB
  float* dec  = (float*)(ws + (3u << 20) + (1u << 19));       // 2.00 MB
  us16*  Wsum = (us16*)(ws + (5u << 20) + (1u << 19));        // 0.25 MB

  pack_all<<<dim3((N_A + N_W + N_E) / 256), 256, 0, stream>>>(
      dec_st, W, enc_in, Abf, Wbf, EinT);
  gemm_dec<<<dim3((TD * BB) / 16, ENC_H / 64), 32, 0, stream>>>(Abf, Wbf, dec);
  softmax_wsum<<<dim3(TD, BB), 256, 0, stream>>>(enc_st, dec, Wsum);
  gemm_out<<<dim3(TD / 16, D_IN / 64, BB), 32, 0, stream>>>(Wsum, EinT, out);
}
